// Attention_80616536146043
// MI455X (gfx1250) — matrix operand order repaired and checked by reading, not compiled, not run on hardware
//
#include <hip/hip_runtime.h>
#include <hip/hip_bf16.h>

// MI455X (gfx1250) fused attention, all matrix math via v_wmma_f32_16x16x32_f16.
//   cvt x f32->f16 ; cvt+transpose weights f32->f16
//   -> QKV GEMM + RoPE (V stored transposed [B,H,D,T])
//   -> flash attention (QK^T + online softmax + P·V)
//   -> output projection -> f32
// All B-operands are kept N-major so every WMMA fragment is two 16-byte loads.
// Global->LDS staging uses CDNA5 async-to-LDS (ASYNCcnt) when available.

typedef _Float16 half_t;
typedef __attribute__((ext_vector_type(16))) _Float16 v16h;
typedef __attribute__((ext_vector_type(8)))  float    v8f;

constexpr int Bv = 2;
constexpr int Tn = 2048;
constexpr int Cn = 1024;
constexpr int Hn = 16;
constexpr int Dn = 64;
constexpr int Mtot = Bv * Tn;   // 4096 rows of x

// ---------------------------------------------------------------------------
// CDNA5 async global->LDS copy (ASYNCcnt path) with safe fallback
// Builtin signature (from compiler diagnostic): params are pointers to
// 4x-int vectors, AS1 (global) source and AS3 (LDS) dest, + imm offset/cpol.
// ---------------------------------------------------------------------------
#if defined(__has_builtin)
#if __has_builtin(__builtin_amdgcn_global_load_async_to_lds_b128)
#define HAVE_ASYNC_LDS 1
#endif
#endif

typedef int v4i __attribute__((vector_size(16)));
typedef __attribute__((address_space(1))) v4i as1_v4i;
typedef __attribute__((address_space(3))) v4i as3_v4i;

// copy 16 bytes global -> LDS
__device__ inline void cp16(half_t* lds, const half_t* g) {
#ifdef HAVE_ASYNC_LDS
  __builtin_amdgcn_global_load_async_to_lds_b128(
      (as1_v4i*)(size_t)g, (as3_v4i*)lds, 0, 0);
#else
  *(uint4*)lds = *(const uint4*)g;
#endif
}

__device__ inline void cp_wait() {
#ifdef HAVE_ASYNC_LDS
#if __has_builtin(__builtin_amdgcn_s_wait_asynccnt)
  __builtin_amdgcn_s_wait_asynccnt(0);
#else
  asm volatile("s_wait_asynccnt 0" ::: "memory");
#endif
#endif
}

// ---------------------------------------------------------------------------
// WMMA fragment helpers (layouts per CDNA5 ISA 7.12.2, wave32)
// ---------------------------------------------------------------------------
__device__ inline v8f wmma16(v16h a, v16h b, v8f c) {
  return __builtin_amdgcn_wmma_f32_16x16x32_f16(
      false, a, false, b, (short)0, c, false, false);
}

__device__ inline v16h frag_2x16(const half_t* c0, const half_t* c1) {
  union { v16h v; uint4 q[2]; } t;
  t.q[0] = *(const uint4*)c0;
  t.q[1] = *(const uint4*)c1;
  return t.v;
}

// A fragment: 16x32 row-major tile, stride ld.
// lane L: row=L&15; K chunks [hf*8 .. +7] and [16+hf*8 .. +7]  (contiguous)
__device__ inline v16h load_A_frag(const half_t* p, int ld) {
  int lane = threadIdx.x & 31;
  const half_t* r = p + (lane & 15) * ld + (lane >> 4) * 8;
  return frag_2x16(r, r + 16);
}

// B fragment from an N-major source (element (k, n) at p[n*ld + k]).
// lane L: col=L&15; K chunks [hf*8 .. +7] and [16+hf*8 .. +7]
__device__ inline v16h load_Bn_frag(const half_t* p, int ld) {
  int lane = threadIdx.x & 31;
  const half_t* r = p + (lane & 15) * ld + (lane >> 4) * 8;
  return frag_2x16(r, r + 16);
}

// ---------------------------------------------------------------------------
// f32 -> f16 conversion (4 elements/thread)
// ---------------------------------------------------------------------------
__global__ void cvt_f32_f16(const float* __restrict__ in,
                            half_t* __restrict__ out, int n) {
  int i = (blockIdx.x * blockDim.x + threadIdx.x) * 4;
  if (i >= n) return;
  float4 f = *(const float4*)(in + i);
  union { half_t h[4]; uint2 u; } t;
  t.h[0] = (half_t)f.x; t.h[1] = (half_t)f.y;
  t.h[2] = (half_t)f.z; t.h[3] = (half_t)f.w;
  *(uint2*)(out + i) = t.u;
}

// f32 [K][N] -> f16 [N][K] transposed (32x32 LDS tiles)
__global__ __launch_bounds__(256) void cvt_transpose_f32_f16(
    const float* __restrict__ in, half_t* __restrict__ out) {
  __shared__ float tl[32][33];
  int k0 = blockIdx.y * 32, n0 = blockIdx.x * 32;
  int r = threadIdx.x >> 5;        // 0..7
  int c = threadIdx.x & 31;
#pragma unroll
  for (int i = 0; i < 4; ++i)
    tl[r + i * 8][c] = in[(size_t)(k0 + r + i * 8) * Cn + n0 + c];
  __syncthreads();
#pragma unroll
  for (int i = 0; i < 4; ++i)
    out[(size_t)(n0 + r + i * 8) * Cn + k0 + c] = (half_t)tl[c][r + i * 8];
}

// ---------------------------------------------------------------------------
// QKV projection: X[4096,1024] x W^T[1024,1024](N-major) -> q/k in [B,H,T,D],
// v in [B,H,D,T] (transposed for P·V fragments).  RoPE on q,k in f32 epilogue.
// 256 threads = 8 waves; tile 128(M) x 128(N) x 32(K);
// wave grid 4(M of 32) x 2(N of 64); 2x4 16x16 WMMA tiles per wave.
// ---------------------------------------------------------------------------
__global__ __launch_bounds__(256) void qkv_rope_gemm(
    const half_t* __restrict__ X,
    const half_t* __restrict__ WqT, const half_t* __restrict__ WkT,
    const half_t* __restrict__ WvT,
    const float* __restrict__ fcos, const float* __restrict__ fsin,
    half_t* __restrict__ Qo, half_t* __restrict__ Ko,
    half_t* __restrict__ Vt) {
  __shared__ __align__(16) unsigned char smem[64 * 1024];
  half_t* As = (half_t*)smem;                    // [128][32]  rows of X
  half_t* Bs = (half_t*)(smem + 128 * 32 * 2);   // [128][32]  rows of W^T
  float*  Cs = (float*)smem;                     // reused: 8 waves x [32][64]

  const int z = blockIdx.z;                      // 0=q 1=k 2=v
  const half_t* WT = (z == 0) ? WqT : (z == 1) ? WkT : WvT;

  const int tid  = threadIdx.x;
  const int wave = tid >> 5;
  const int lane = tid & 31;
  const int wm = wave & 3, wn = wave >> 2;
  const int m0 = blockIdx.y * 128;
  const int n0 = blockIdx.x * 128;
  const int col = lane & 15, hf = lane >> 4;

  v8f acc[2][4] = {};

  const int row = tid >> 1;                 // 0..127
  const int cs  = (tid & 1) * 16;           // 0 or 16

  for (int k0 = 0; k0 < Cn; k0 += 32) {
    const half_t* sa = X  + (size_t)(m0 + row) * Cn + k0 + cs;
    const half_t* sb = WT + (size_t)(n0 + row) * Cn + k0 + cs;
    half_t* da = As + row * 32 + cs;
    half_t* db = Bs + row * 32 + cs;
    cp16(da, sa);       cp16(da + 8, sa + 8);
    cp16(db, sb);       cp16(db + 8, sb + 8);
    cp_wait();
    __syncthreads();

    v16h bfrag[4];
#pragma unroll
    for (int ni = 0; ni < 4; ++ni)
      bfrag[ni] = load_Bn_frag(Bs + (wn * 64 + ni * 16) * 32, 32);
#pragma unroll
    for (int mi = 0; mi < 2; ++mi) {
      v16h a = load_A_frag(As + (wm * 32 + mi * 16) * 32, 32);
#pragma unroll
      for (int ni = 0; ni < 4; ++ni)
        acc[mi][ni] = wmma16(a, bfrag[ni], acc[mi][ni]);
    }
    __syncthreads();
  }

  // spill f32 tile to LDS so RoPE pairs (d, d+1) are per-lane contiguous
  float* ct = Cs + wave * (32 * 64);
#pragma unroll
  for (int mi = 0; mi < 2; ++mi)
#pragma unroll
    for (int ni = 0; ni < 4; ++ni)
#pragma unroll
      for (int v = 0; v < 8; ++v)
        ct[(mi * 16 + v + 8 * hf) * 64 + ni * 16 + col] = acc[mi][ni][v];

  const int h = (n0 + wn * 64) / Dn;        // 64-aligned -> single head
#pragma unroll
  for (int p = 0; p < 32; ++p) {
    int idx = p * 32 + lane;                // 1024 pairs in the 32x64 tile
    int r   = idx >> 5;
    int pc  = idx & 31;                     // pair column (d/2)
    int gm  = m0 + wm * 32 + r;
    int b   = gm / Tn, t = gm % Tn;
    float a0 = ct[r * 64 + 2 * pc];
    float a1 = ct[r * 64 + 2 * pc + 1];
    if (z < 2) {
      float c = fcos[t * (Dn / 2) + pc];
      float s = fsin[t * (Dn / 2) + pc];
      float e0 = a0 * c - a1 * s;
      float e1 = a0 * s + a1 * c;
      half_t* Out = (z == 0) ? Qo : Ko;     // [B,H,T,D]
      size_t o = (((size_t)(b * Hn + h) * Tn + t) * Dn) + 2 * pc;
      Out[o]     = (half_t)e0;
      Out[o + 1] = (half_t)e1;
    } else {                                // V -> [B,H,D,T]
      size_t o = (((size_t)(b * Hn + h) * Dn) + 2 * pc) * Tn + t;
      Vt[o]      = (half_t)a0;
      Vt[o + Tn] = (half_t)a1;
    }
  }
}

// ---------------------------------------------------------------------------
// Flash attention: one 16-row Q tile per wave, 32-key blocks, online softmax.
// S = Q·K^T (4 WMMAs), P·V (4 WMMAs) per key block.  Output [B,T,C] f16.
// ---------------------------------------------------------------------------
__global__ __launch_bounds__(256) void flash_attn(
    const half_t* __restrict__ Q, const half_t* __restrict__ K,
    const half_t* __restrict__ Vt, half_t* __restrict__ O) {
  __shared__ __align__(16) half_t Pl[8][16 * 32];

  const int tid = threadIdx.x, wave = tid >> 5, lane = tid & 31;
  const int col = lane & 15, hf = lane >> 4;
  const int gt = blockIdx.x * 8 + wave;
  const int bh = gt >> 7;              // / (T/16)
  const int qt = gt & 127;

  const half_t* qp = Q + ((size_t)bh * Tn + qt * 16) * Dn;
  v16h aq0 = load_A_frag(qp, Dn);        // k = 0..31
  v16h aq1 = load_A_frag(qp + 32, Dn);   // k = 32..63

  v8f o[4] = {};
  float mrow[8], lrow[8];
#pragma unroll
  for (int v = 0; v < 8; ++v) { mrow[v] = -1e30f; lrow[v] = 0.f; }

  const int nj = (qt * 16 + 15) / 32 + 1;
  half_t* pl = Pl[wave];

  for (int jb = 0; jb < nj; ++jb) {
    const half_t* kp = K + ((size_t)bh * Tn + jb * 32) * Dn;   // keys N-major
    v8f s0 = {}, s1 = {};
    s0 = wmma16(aq0, load_Bn_frag(kp, Dn), s0);
    s0 = wmma16(aq1, load_Bn_frag(kp + 32, Dn), s0);
    s1 = wmma16(aq0, load_Bn_frag(kp + 16 * Dn, Dn), s1);
    s1 = wmma16(aq1, load_Bn_frag(kp + 16 * Dn + 32, Dn), s1);

#pragma unroll
    for (int v = 0; v < 8; ++v) {
      int rowi = v + 8 * hf;
      int qrow = qt * 16 + rowi;
      float x0 = s0[v] * 0.125f;           // 1/sqrt(64)
      float x1 = s1[v] * 0.125f;
      if (jb * 32 + col > qrow)      x0 = -1e30f;
      if (jb * 32 + 16 + col > qrow) x1 = -1e30f;
      float pm = fmaxf(x0, x1);
#pragma unroll
      for (int d = 1; d < 16; d <<= 1) pm = fmaxf(pm, __shfl_xor(pm, d, 32));
      float mn    = fmaxf(mrow[v], pm);
      float alpha = __expf(mrow[v] - mn);
      float p0 = __expf(x0 - mn);
      float p1 = __expf(x1 - mn);
      float ps = p0 + p1;
#pragma unroll
      for (int d = 1; d < 16; d <<= 1) ps += __shfl_xor(ps, d, 32);
      lrow[v] = lrow[v] * alpha + ps;
      mrow[v] = mn;
#pragma unroll
      for (int n = 0; n < 4; ++n) o[n][v] = o[n][v] * alpha;
      pl[rowi * 32 + col]      = (half_t)p0;
      pl[rowi * 32 + 16 + col] = (half_t)p1;
    }

    v16h pa = load_A_frag(pl, 32);        // wave-local LDS bounce
#pragma unroll
    for (int n = 0; n < 4; ++n) {
      const half_t* vp = Vt + ((size_t)bh * Dn + n * 16) * Tn + jb * 32;
      o[n] = wmma16(pa, load_Bn_frag(vp, Tn), o[n]);
    }
  }

  const int b = bh / Hn, h = bh % Hn;
#pragma unroll
  for (int n = 0; n < 4; ++n)
#pragma unroll
    for (int v = 0; v < 8; ++v) {
      int t = qt * 16 + v + 8 * hf;
      float val = o[n][v] / lrow[v];
      O[((size_t)(b * Tn + t)) * Cn + h * Dn + n * 16 + col] = (half_t)val;
    }
}

// ---------------------------------------------------------------------------
// Output projection: attn[4096,1024] x Wo^T(N-major) -> f32 out.
// ---------------------------------------------------------------------------
__global__ __launch_bounds__(256) void out_proj_gemm(
    const half_t* __restrict__ A, const half_t* __restrict__ WoT,
    float* __restrict__ Out) {
  __shared__ __align__(16) half_t As[128 * 32];
  __shared__ __align__(16) half_t Bs[128 * 32];

  const int tid  = threadIdx.x;
  const int wave = tid >> 5;
  const int lane = tid & 31;
  const int wm = wave & 3, wn = wave >> 2;
  const int m0 = blockIdx.y * 128;
  const int n0 = blockIdx.x * 128;
  const int col = lane & 15, hf = lane >> 4;

  v8f acc[2][4] = {};

  const int row = tid >> 1;
  const int cs  = (tid & 1) * 16;

  for (int k0 = 0; k0 < Cn; k0 += 32) {
    const half_t* sa = A   + (size_t)(m0 + row) * Cn + k0 + cs;
    const half_t* sb = WoT + (size_t)(n0 + row) * Cn + k0 + cs;
    half_t* da = As + row * 32 + cs;
    half_t* db = Bs + row * 32 + cs;
    cp16(da, sa);       cp16(da + 8, sa + 8);
    cp16(db, sb);       cp16(db + 8, sb + 8);
    cp_wait();
    __syncthreads();

    v16h bfrag[4];
#pragma unroll
    for (int ni = 0; ni < 4; ++ni)
      bfrag[ni] = load_Bn_frag(Bs + (wn * 64 + ni * 16) * 32, 32);
#pragma unroll
    for (int mi = 0; mi < 2; ++mi) {
      v16h a = load_A_frag(As + (wm * 32 + mi * 16) * 32, 32);
#pragma unroll
      for (int ni = 0; ni < 4; ++ni)
        acc[mi][ni] = wmma16(a, bfrag[ni], acc[mi][ni]);
    }
    __syncthreads();
  }

#pragma unroll
  for (int mi = 0; mi < 2; ++mi)
#pragma unroll
    for (int ni = 0; ni < 4; ++ni)
#pragma unroll
      for (int v = 0; v < 8; ++v) {
        size_t gm = m0 + wm * 32 + mi * 16 + v + 8 * hf;
        Out[gm * Cn + n0 + wn * 64 + ni * 16 + col] = acc[mi][ni][v];
      }
}

// ---------------------------------------------------------------------------
// host launcher
// ---------------------------------------------------------------------------
extern "C" void kernel_launch(void* const* d_in, const int* in_sizes, int n_in,
                              void* d_out, int out_size, void* d_ws,
                              size_t ws_size, hipStream_t stream) {
  (void)in_sizes; (void)n_in; (void)out_size; (void)ws_size;

  const float* x  = (const float*)d_in[0];
  const float* fc = (const float*)d_in[1];
  const float* fs = (const float*)d_in[2];
  const float* Wq = (const float*)d_in[3];
  const float* Wk = (const float*)d_in[4];
  const float* Wv = (const float*)d_in[5];
  const float* Wo = (const float*)d_in[6];
  float* out = (float*)d_out;

  char* ws = (char*)d_ws;
  size_t off = 0;
  auto carve = [&](size_t bytes) -> void* {
    void* p = ws + off;
    off += (bytes + 255) & ~(size_t)255;
    return p;
  };
  const size_t nX = (size_t)Mtot * Cn;
  const size_t nW = (size_t)Cn * Cn;
  const size_t nH = (size_t)Bv * Hn * Tn * Dn;

  half_t* xb   = (half_t*)carve(nX * 2);
  half_t* wqT  = (half_t*)carve(nW * 2);
  half_t* wkT  = (half_t*)carve(nW * 2);
  half_t* wvT  = (half_t*)carve(nW * 2);
  half_t* woT  = (half_t*)carve(nW * 2);
  half_t* qb   = (half_t*)carve(nH * 2);   // [B,H,T,D]
  half_t* kb   = (half_t*)carve(nH * 2);   // [B,H,T,D]
  half_t* vbt  = (half_t*)carve(nH * 2);   // [B,H,D,T]
  half_t* ab   = (half_t*)carve(nX * 2);   // [B,T,C]

  cvt_f32_f16<<<(int)(nX / 1024), 256, 0, stream>>>(x, xb, (int)nX);
  dim3 tg(Cn / 32, Cn / 32);
  cvt_transpose_f32_f16<<<tg, 256, 0, stream>>>(Wq, wqT);
  cvt_transpose_f32_f16<<<tg, 256, 0, stream>>>(Wk, wkT);
  cvt_transpose_f32_f16<<<tg, 256, 0, stream>>>(Wv, wvT);
  cvt_transpose_f32_f16<<<tg, 256, 0, stream>>>(Wo, woT);

  qkv_rope_gemm<<<dim3(Cn / 128, Mtot / 128, 3), 256, 0, stream>>>(
      xb, wqT, wkT, wvT, fc, fs, qb, kb, vbt);

  flash_attn<<<dim3(Bv * Hn * (Tn / 16) / 8), 256, 0, stream>>>(qb, kb, vbt, ab);

  out_proj_gemm<<<dim3(Cn / 128, Mtot / 128), 256, 0, stream>>>(ab, woT, out);
}
